// BiomarkerGNN_45973329936474
// MI455X (gfx1250) — matrix operand order repaired and checked by reading, not compiled, not run on hardware
//
#include <hip/hip_runtime.h>
#include <hip/hip_bf16.h>
#include <math.h>

// ---------------------------------------------------------------------------
// BiomarkerGNN forward for MI455X (gfx1250, wave32).
//   - All dense GEMMs run on v_wmma_f32_16x16x32_f16 (f16 in, f32 accum).
//   - Weights pre-transposed+converted to f16 once per call, staged in LDS
//     with a padded stride (K+8 halves) to avoid 16-way bank conflicts.
//   - B fragments for a K-chunk are batch-loaded so the 4 WMMAs issue
//     back-to-back behind a single s_wait_dscnt.
//   - Edge aggregation = global_atomic_add_f32 segment sum (L2-resident:
//     the 25MB activation table fits in the 192MB L2).
// Workspace requirement: ~142 MB (see layout in kernel_launch).
// ---------------------------------------------------------------------------

#define N_NODES 100000
#define N_EDGES 1200000
#define N_FEAT  128
#define HID     64
#define PROJ    32

typedef _Float16 v8h  __attribute__((ext_vector_type(8)));
typedef _Float16 v16h __attribute__((ext_vector_type(16)));
typedef float    v8f  __attribute__((ext_vector_type(8)));

// ---------------- small utility kernels ----------------

__global__ void pack_wt_kernel(const float* __restrict__ W, _Float16* __restrict__ Wt,
                               int K, int M) {
    int idx = blockIdx.x * blockDim.x + threadIdx.x;
    if (idx < K * M) {
        int k = idx / M, m = idx % M;
        Wt[(size_t)m * K + k] = (_Float16)W[idx];   // Wt = W^T, f16
    }
}

__global__ void bn_prep_kernel(const float* __restrict__ b1, const float* __restrict__ gamma,
                               const float* __restrict__ beta, const float* __restrict__ rmean,
                               const float* __restrict__ rvar,
                               float* __restrict__ sc, float* __restrict__ sh) {
    int f = threadIdx.x;
    if (f < HID) {
        float s = gamma[f] * rsqrtf(rvar[f] + 1e-5f);
        sc[f] = s;
        sh[f] = (b1[f] - rmean[f]) * s + beta[f];
    }
}

__global__ void f32_to_f16_kernel(const float* __restrict__ x, _Float16* __restrict__ xh, int n) {
    int i = blockIdx.x * blockDim.x + threadIdx.x;
    if (i < n) xh[i] = (_Float16)x[i];
}

__global__ void zero_f32_kernel(float* __restrict__ p, int n) {
    int i = blockIdx.x * blockDim.x + threadIdx.x;
    if (i < n) p[i] = 0.0f;
}

__global__ void deg_init_kernel(float* __restrict__ deg, int n) {
    int i = blockIdx.x * blockDim.x + threadIdx.x;
    if (i < n) deg[i] = 1.0f;                      // self loop
}

__global__ void deg_accum_kernel(const int* __restrict__ ei, float* __restrict__ deg, int E) {
    int e = blockIdx.x * blockDim.x + threadIdx.x;
    if (e < E) atomicAdd(&deg[ei[E + e]], 1.0f);   // dst row
}

__global__ void deg_finish_kernel(float* __restrict__ dinv, int n) {
    int i = blockIdx.x * blockDim.x + threadIdx.x;
    if (i < n) dinv[i] = rsqrtf(dinv[i]);
}

// ---------------- WMMA GEMM: C[N,M] = A[N,K] @ Wt^T, fused affine epilogue ----

template <int K, int M>
__global__ __launch_bounds__(256) void gemm_wmma_kernel(
    const _Float16* __restrict__ A, const _Float16* __restrict__ Wt,
    float* __restrict__ C, _Float16* __restrict__ Ch,
    const float* __restrict__ scale, const float* __restrict__ shift,
    int N, int do_relu) {
    constexpr int LDSK = K + 8;                    // pad: stride 68 dwords, no bank conflict
    constexpr int MT = M / 16;
    __shared__ _Float16 w[M * LDSK];

    // cooperative weight stage: Wt is [M][K] f16 row-major
    for (int i = threadIdx.x; i < (M * K) / 8; i += 256) {
        int m = i / (K / 8);
        int kk = (i % (K / 8)) * 8;
        *(v8h*)&w[m * LDSK + kk] = *(const v8h*)&Wt[(size_t)m * K + kk];
    }
    __syncthreads();

    const int lane = threadIdx.x & 31;
    const int wv   = threadIdx.x >> 5;             // 8 waves per block
    const int half = lane >> 4;                    // K-half select
    const int l16  = lane & 15;

    const int row  = blockIdx.x * 128 + wv * 16 + l16;
    const int arow = row < N ? row : N - 1;        // clamp for tail tile
    const _Float16* ap = A + (size_t)arow * K + half * 8;

    v8f zero8 = {0.f, 0.f, 0.f, 0.f, 0.f, 0.f, 0.f, 0.f};
    v8f acc[MT];
#pragma unroll
    for (int c = 0; c < MT; ++c) acc[c] = zero8;

#pragma unroll
    for (int k0 = 0; k0 < K; k0 += 32) {
        // A fragment 16x32 f16: lanes<16 -> K {k0..k0+7, k0+16..k0+23}; lanes>=16 -> +8
        v8h alo = *(const v8h*)(ap + k0);
        v8h ahi = *(const v8h*)(ap + k0 + 16);
        v16h a;
#pragma unroll
        for (int i = 0; i < 8; ++i) { a[i] = alo[i]; a[8 + i] = ahi[i]; }

        // Batch-load all B fragments for this K-chunk (independent ds clauses),
        // then issue the MT WMMAs back-to-back behind a single dscnt wait.
        v16h b[MT];
#pragma unroll
        for (int c = 0; c < MT; ++c) {
            // B fragment 32x16 f16: lane holds col n = l16; lanes<16 -> K {k0..k0+7,
            // k0+16..k0+23}; lanes>=16 -> +8 (matches A)
            const _Float16* bp = &w[(c * 16 + l16) * LDSK + k0 + half * 8];
            v8h blo = *(const v8h*)bp;
            v8h bhi = *(const v8h*)(bp + 16);
#pragma unroll
            for (int i = 0; i < 8; ++i) { b[c][i] = blo[i]; b[c][8 + i] = bhi[i]; }
        }
#pragma unroll
        for (int c = 0; c < MT; ++c) {
            acc[c] = __builtin_amdgcn_wmma_f32_16x16x32_f16(
                false, a, false, b[c], (short)0, acc[c], false, false);
        }
    }

    // C/D layout: lane L -> col = L%16; VGPR v -> row = v + 8*(L/16)
#pragma unroll
    for (int c = 0; c < MT; ++c) {
        int col = c * 16 + l16;
        float s = scale ? scale[col] : 1.0f;
        float t = shift ? shift[col] : 0.0f;
#pragma unroll
        for (int v = 0; v < 8; ++v) {
            int r = blockIdx.x * 128 + wv * 16 + half * 8 + v;
            if (r < N) {
                float y = acc[c][v] * s + t;
                if (do_relu) y = fmaxf(y, 0.0f);
                if (C)  C[(size_t)r * M + col] = y;
                if (Ch) Ch[(size_t)r * M + col] = (_Float16)y;
            }
        }
    }
}

// ---------------- GCN edge aggregation ----------------

__global__ void gcn_agg_kernel(const int* __restrict__ ei, const float* __restrict__ dinv,
                               const float* __restrict__ hw, float* __restrict__ agg, int E) {
    int gid = blockIdx.x * blockDim.x + threadIdx.x;   // E*32 threads
    int e = gid >> 5, f = gid & 31;
    if (e >= E) return;
    int s = ei[e], d = ei[E + e];
    float nw = dinv[s] * dinv[d];
    size_t sb = (size_t)s * HID, db = (size_t)d * HID;
    atomicAdd(&agg[db + f],      hw[sb + f]      * nw);
    atomicAdd(&agg[db + f + 32], hw[sb + f + 32] * nw);
}

// hg = relu(agg + hw*dinv^2 + b); also emit f16 copy. hg may alias agg (same index).
__global__ void gcn_finalize_kernel(const float* agg, const float* __restrict__ hw,
                                    const float* __restrict__ dinv, const float* __restrict__ b,
                                    float* hg, _Float16* __restrict__ hg_h, int N) {
    int gid = blockIdx.x * blockDim.x + threadIdx.x;   // N*64 threads
    int i = gid >> 6, f = gid & 63;
    if (i >= N) return;
    float di = dinv[i];
    size_t idx = (size_t)i * HID + f;
    float v = agg[idx] + hw[idx] * di * di + b[f];
    v = fmaxf(v, 0.0f);
    hg[idx] = v;
    hg_h[idx] = (_Float16)v;
}

// ---------------- gate + fusion + classifier (one wave per node) ----------------

__global__ __launch_bounds__(256) void gate_fuse_kernel(
    const float* __restrict__ hmlp, const float* __restrict__ hg,
    const float* __restrict__ gateW, const float* __restrict__ gateb,
    const float* __restrict__ cW, const float* __restrict__ cb,
    _Float16* __restrict__ hfused_h, float* __restrict__ logits, int N) {
    int node = blockIdx.x * 8 + (threadIdx.x >> 5);
    int lane = threadIdx.x & 31;
    if (node >= N) return;                             // wave-uniform exit
    size_t base = (size_t)node * HID;
    float m0 = hmlp[base + lane], m1 = hmlp[base + lane + 32];
    float g0 = hg[base + lane],   g1 = hg[base + lane + 32];
    float p = m0 * gateW[lane] + m1 * gateW[lane + 32] +
              g0 * gateW[64 + lane] + g1 * gateW[96 + lane];
#pragma unroll
    for (int o = 16; o > 0; o >>= 1) p += __shfl_xor(p, o, 32);
    float gate = 1.0f / (1.0f + expf(-(p + gateb[0])));
    float f0 = gate * g0 + (1.0f - gate) * m0;
    float f1 = gate * g1 + (1.0f - gate) * m1;
    hfused_h[base + lane]      = (_Float16)f0;
    hfused_h[base + lane + 32] = (_Float16)f1;
    float q = f0 * cW[lane] + f1 * cW[lane + 32];
#pragma unroll
    for (int o = 16; o > 0; o >>= 1) q += __shfl_xor(q, o, 32);
    if (lane == 0) logits[node] = q + cb[0];
}

// ---------------- host orchestration ----------------

static inline size_t align256(size_t x) { return (x + 255) & ~(size_t)255; }

extern "C" void kernel_launch(void* const* d_in, const int* in_sizes, int n_in,
                              void* d_out, int out_size, void* d_ws, size_t ws_size,
                              hipStream_t stream) {
    const float* x     = (const float*)d_in[0];
    const int*   ei    = (const int*)d_in[1];
    const float* W1    = (const float*)d_in[2];
    const float* b1    = (const float*)d_in[3];
    const float* gamma = (const float*)d_in[4];
    const float* beta  = (const float*)d_in[5];
    const float* rmean = (const float*)d_in[6];
    const float* rvar  = (const float*)d_in[7];
    const float* gW0   = (const float*)d_in[8];
    const float* gb0   = (const float*)d_in[9];
    const float* gW1   = (const float*)d_in[10];
    const float* gb1   = (const float*)d_in[11];
    const float* gW2   = (const float*)d_in[12];
    const float* gb2   = (const float*)d_in[13];
    const float* gateW = (const float*)d_in[14];
    const float* gateb = (const float*)d_in[15];
    const float* pW1   = (const float*)d_in[16];
    const float* pb1   = (const float*)d_in[17];
    const float* pW2   = (const float*)d_in[18];
    const float* pb2   = (const float*)d_in[19];
    const float* cW    = (const float*)d_in[20];
    const float* cb    = (const float*)d_in[21];
    float* out = (float*)d_out;

    const int N = N_NODES, E = N_EDGES;

    // workspace carve-up
    char* p = (char*)d_ws;
    size_t off = 0;
    auto carve = [&](size_t bytes) { void* r = p + off; off = align256(off + bytes); return r; };
    float*     dinv  = (float*)carve((size_t)N * 4);
    _Float16*  xh    = (_Float16*)carve((size_t)N * N_FEAT * 2);
    float*     hmlp  = (float*)carve((size_t)N * HID * 4);
    float*     hwb   = (float*)carve((size_t)N * HID * 4);
    float*     aggb  = (float*)carve((size_t)N * HID * 4);
    _Float16*  hgh   = (_Float16*)carve((size_t)N * HID * 2);
    _Float16*  hfh   = (_Float16*)carve((size_t)N * HID * 2);
    _Float16*  t1h   = (_Float16*)carve((size_t)N * HID * 2);
    _Float16*  W1t   = (_Float16*)carve((size_t)N_FEAT * HID * 2);
    _Float16*  gW0t  = (_Float16*)carve((size_t)N_FEAT * HID * 2);
    _Float16*  gW1t  = (_Float16*)carve((size_t)HID * HID * 2);
    _Float16*  gW2t  = (_Float16*)carve((size_t)HID * HID * 2);
    _Float16*  pW1t  = (_Float16*)carve((size_t)HID * HID * 2);
    _Float16*  pW2t  = (_Float16*)carve((size_t)HID * PROJ * 2);
    float*     bnsc  = (float*)carve(HID * 4);
    float*     bnsh  = (float*)carve(HID * 4);
    (void)ws_size; (void)in_sizes; (void)n_in; (void)out_size;

    const int T = 256;
    const int gemm_blocks = (N + 127) / 128;

    // 1) pack weights f16-transposed, BN affine prep
    pack_wt_kernel<<<(N_FEAT * HID + T - 1) / T, T, 0, stream>>>(W1,  W1t,  N_FEAT, HID);
    pack_wt_kernel<<<(N_FEAT * HID + T - 1) / T, T, 0, stream>>>(gW0, gW0t, N_FEAT, HID);
    pack_wt_kernel<<<(HID * HID + T - 1) / T, T, 0, stream>>>(gW1, gW1t, HID, HID);
    pack_wt_kernel<<<(HID * HID + T - 1) / T, T, 0, stream>>>(gW2, gW2t, HID, HID);
    pack_wt_kernel<<<(HID * HID + T - 1) / T, T, 0, stream>>>(pW1, pW1t, HID, HID);
    pack_wt_kernel<<<(HID * PROJ + T - 1) / T, T, 0, stream>>>(pW2, pW2t, HID, PROJ);
    bn_prep_kernel<<<1, 64, 0, stream>>>(b1, gamma, beta, rmean, rvar, bnsc, bnsh);

    // 2) degrees -> dinv
    deg_init_kernel<<<(N + T - 1) / T, T, 0, stream>>>(dinv, N);
    deg_accum_kernel<<<(E + T - 1) / T, T, 0, stream>>>(ei, dinv, E);
    deg_finish_kernel<<<(N + T - 1) / T, T, 0, stream>>>(dinv, N);

    // 3) x -> f16
    f32_to_f16_kernel<<<(N * N_FEAT + T - 1) / T, T, 0, stream>>>(x, xh, N * N_FEAT);

    // 4) MLP branch: h_mlp = relu(BN(x@W1 + b1))  (fused affine epilogue)
    gemm_wmma_kernel<N_FEAT, HID><<<gemm_blocks, T, 0, stream>>>(
        xh, W1t, hmlp, nullptr, bnsc, bnsh, N, 1);

    // 5) GCN layer 0
    gemm_wmma_kernel<N_FEAT, HID><<<gemm_blocks, T, 0, stream>>>(
        xh, gW0t, hwb, nullptr, nullptr, nullptr, N, 0);
    zero_f32_kernel<<<(N * HID + T - 1) / T, T, 0, stream>>>(aggb, N * HID);
    gcn_agg_kernel<<<(E * 32 + T - 1) / T, T, 0, stream>>>(ei, dinv, hwb, aggb, E);
    gcn_finalize_kernel<<<(N * HID + T - 1) / T, T, 0, stream>>>(aggb, hwb, dinv, gb0, aggb, hgh, N);

    // 6) GCN layer 1
    gemm_wmma_kernel<HID, HID><<<gemm_blocks, T, 0, stream>>>(
        hgh, gW1t, hwb, nullptr, nullptr, nullptr, N, 0);
    zero_f32_kernel<<<(N * HID + T - 1) / T, T, 0, stream>>>(aggb, N * HID);
    gcn_agg_kernel<<<(E * 32 + T - 1) / T, T, 0, stream>>>(ei, dinv, hwb, aggb, E);
    gcn_finalize_kernel<<<(N * HID + T - 1) / T, T, 0, stream>>>(aggb, hwb, dinv, gb1, aggb, hgh, N);

    // 7) GCN layer 2
    gemm_wmma_kernel<HID, HID><<<gemm_blocks, T, 0, stream>>>(
        hgh, gW2t, hwb, nullptr, nullptr, nullptr, N, 0);
    zero_f32_kernel<<<(N * HID + T - 1) / T, T, 0, stream>>>(aggb, N * HID);
    gcn_agg_kernel<<<(E * 32 + T - 1) / T, T, 0, stream>>>(ei, dinv, hwb, aggb, E);
    gcn_finalize_kernel<<<(N * HID + T - 1) / T, T, 0, stream>>>(aggb, hwb, dinv, gb2, aggb, hgh, N);

    // 8) gate + fusion + classifier: logits -> out[0:N), h_fused -> f16
    gate_fuse_kernel<<<(N + 7) / 8, T, 0, stream>>>(
        hmlp, aggb, gateW, gateb, cW, cb, hfh, out, N);

    // 9) projection head: t1 = relu(h_fused@pW1 + pb1); z = t1@pW2 + pb2 -> out[N:]
    gemm_wmma_kernel<HID, HID><<<gemm_blocks, T, 0, stream>>>(
        hfh, pW1t, nullptr, t1h, nullptr, pb1, N, 1);
    gemm_wmma_kernel<HID, PROJ><<<gemm_blocks, T, 0, stream>>>(
        t1h, pW2t, out + N, nullptr, nullptr, pb2, N, 0);
}
